// CausalSelfAttention_15504831939088
// MI455X (gfx1250) — matrix operand order repaired and checked by reading, not compiled, not run on hardware
//
#include <hip/hip_runtime.h>
#include <hip/hip_bf16.h>

typedef _Float16 v16h __attribute__((ext_vector_type(16)));
typedef _Float16 v8h  __attribute__((ext_vector_type(8)));
typedef float    v8f  __attribute__((ext_vector_type(8)));

#define ATTN_B  2
#define ATTN_S  2048
#define ATTN_D  1024
#define ATTN_H  16
#define ATTN_DH 64
#define ATTN_M  (ATTN_B * ATTN_S)   // 4096 rows for projections

// ---------------------------------------------------------------------------
// CDNA5 async global->LDS copy (ASYNCcnt-tracked), 16B per lane.
// the immediate offset is added to both the LDS and the global address
// Async loads retire in order per wave -> partial waits enable double buffering.
// ---------------------------------------------------------------------------
template <int OFF>
__device__ __forceinline__ void async_b128(const _Float16* g, _Float16* l) {
    const unsigned lds = (unsigned)(uintptr_t)l;   // low 32b of LDS aperture addr
    asm volatile("global_load_async_to_lds_b128 %0, %1, off offset:%2"
                 :: "v"(lds), "v"(g), "i"(OFF) : "memory");
}
template <int N>
__device__ __forceinline__ void wait_async_le() {
    asm volatile("s_wait_asynccnt %0" :: "i"(N) : "memory");
}

// ---------------------------------------------------------------------------
// fp32 -> f16 conversion of x (vectorized)
// ---------------------------------------------------------------------------
__global__ void cvt_x_kernel(const float* __restrict__ x, _Float16* __restrict__ xh, int n4) {
    int i = blockIdx.x * blockDim.x + threadIdx.x;
    if (i >= n4) return;
    float4 v = ((const float4*)x)[i];
    xh[i * 4 + 0] = (_Float16)v.x;
    xh[i * 4 + 1] = (_Float16)v.y;
    xh[i * 4 + 2] = (_Float16)v.z;
    xh[i * 4 + 3] = (_Float16)v.w;
}

// ---------------------------------------------------------------------------
// fp32 W[K][N] -> f16 Wt[N][K] (LDS tile transpose, 32x32 tiles)
// ---------------------------------------------------------------------------
__global__ void cvt_tr_kernel(const float* __restrict__ W, _Float16* __restrict__ Wt) {
    __shared__ float tile[32][33];
    const int bx = blockIdx.x * 32, by = blockIdx.y * 32;
    const int tx = threadIdx.x, ty = threadIdx.y;
    for (int i = 0; i < 32; i += 8)
        tile[ty + i][tx] = W[(size_t)(by + ty + i) * ATTN_D + bx + tx];
    __syncthreads();
    for (int i = 0; i < 32; i += 8)
        Wt[(size_t)(bx + ty + i) * ATTN_D + by + tx] = (_Float16)tile[tx][ty + i];
}

// ---------------------------------------------------------------------------
// WMMA GEMM: C[4096][1024] = A[4096][1024](f16) * Bt[1024(n)][1024(k)](f16) + bias
// mode 0: write f16 into [B][H][S][DH]   (Q / K layout)
// mode 1: write f16 into [B][H][DH][S]   (V transposed layout)
// mode 3: write f32 row-major [M][N]     (final output + bias)
// Block: 128 threads = 4 waves; block tile 64x64; per-wave 16(m) x 64(n).
// Double-buffered GLOBAL_LOAD_ASYNC_TO_LDS_B128 staging.
// ---------------------------------------------------------------------------
__global__ __launch_bounds__(128) void gemm_wmma_kernel(
    const _Float16* __restrict__ A, const _Float16* __restrict__ Bt,
    const float* __restrict__ bias, void* __restrict__ dstv, int mode)
{
    constexpr int Kd = ATTN_D;
    const int m0 = blockIdx.x * 64;
    const int n0 = blockIdx.y * 64;
    const int lane = threadIdx.x & 31;
    const int w    = threadIdx.x >> 5;
    const int l15  = lane & 15;
    const int hi   = (lane & 16) ? 1 : 0;
    const int koff  = hi ? 8 : 0;   // A-fragment K sub-offset
    const int kbase = hi ? 16 : 0;  // unused; B fragment also uses koff
    const int rowb  = hi ? 8 : 0;   // C/D row base

    __shared__ _Float16 As[2][64][40]; // double-buffered 64 m-rows x 32 k (+pad)
    __shared__ _Float16 Bs[2][64][40]; // double-buffered 64 n-rows x 32 k (+pad)

    v8f acc[4] = {};

    const int ldr = threadIdx.x >> 1;        // 0..63
    const int ldc = (threadIdx.x & 1) * 16;  // 0 or 16 halfs

    const _Float16* gA = A  + (size_t)(m0 + ldr) * Kd + ldc;
    const _Float16* gB = Bt + (size_t)(n0 + ldr) * Kd + ldc;

    // prologue: stage tile 0 into buffer 0 (4 async copies per thread)
    async_b128<0>(gA, &As[0][ldr][ldc]);  async_b128<16>(gA, &As[0][ldr][ldc]);
    async_b128<0>(gB, &Bs[0][ldr][ldc]);  async_b128<16>(gB, &Bs[0][ldr][ldc]);

    int buf = 0;
    for (int k0 = 0; k0 < Kd; k0 += 32, buf ^= 1) {
        if (k0 + 32 < Kd) {
            // stage next tile into the other buffer, then retire current (<=4)
            const _Float16* na = gA + k0 + 32;
            const _Float16* nb = gB + k0 + 32;
            async_b128<0>(na, &As[buf ^ 1][ldr][ldc]);
            async_b128<16>(na, &As[buf ^ 1][ldr][ldc]);
            async_b128<0>(nb, &Bs[buf ^ 1][ldr][ldc]);
            async_b128<16>(nb, &Bs[buf ^ 1][ldr][ldc]);
            __builtin_prefetch(na + 32, 0, 1);   // global_prefetch_b8, tile k+2
            __builtin_prefetch(nb + 32, 0, 1);
            wait_async_le<4>();
        } else {
            wait_async_le<0>();
        }
        __syncthreads();

        v16h af;
        ((v8h*)&af)[0] = *(const v8h*)&As[buf][w * 16 + l15][koff];
        ((v8h*)&af)[1] = *(const v8h*)&As[buf][w * 16 + l15][koff + 16];
        for (int t = 0; t < 4; ++t) {
            v16h bf;
            ((v8h*)&bf)[0] = *(const v8h*)&Bs[buf][t * 16 + l15][koff];
            ((v8h*)&bf)[1] = *(const v8h*)&Bs[buf][t * 16 + l15][koff + 16];
            acc[t] = __builtin_amdgcn_wmma_f32_16x16x32_f16(
                false, af, false, bf, (short)0, acc[t], false, false);
        }
        __syncthreads();
    }

    for (int t = 0; t < 4; ++t) {
        const int n = n0 + t * 16 + l15;
        const float bval = bias[n];
        for (int r = 0; r < 8; ++r) {
            const int m = m0 + w * 16 + rowb + r;
            const float v = acc[t][r] + bval;
            if (mode == 0) {
                const int b = m >> 11, s = m & (ATTN_S - 1);
                const int h = n >> 6, dh = n & (ATTN_DH - 1);
                ((_Float16*)dstv)[(((size_t)(b * ATTN_H + h) * ATTN_S) + s) * ATTN_DH + dh] = (_Float16)v;
            } else if (mode == 1) {
                const int b = m >> 11, s = m & (ATTN_S - 1);
                const int h = n >> 6, dh = n & (ATTN_DH - 1);
                ((_Float16*)dstv)[(((size_t)(b * ATTN_H + h) * ATTN_DH) + dh) * ATTN_S + s] = (_Float16)v;
            } else {
                ((float*)dstv)[(size_t)m * ATTN_D + n] = v;
            }
        }
    }
}

// ---------------------------------------------------------------------------
// Flash attention: per block one (b,h) and 64 q-rows. 4 waves x 16 rows.
// Q,K: [B][H][S][DH] f16;  Vt: [B][H][DH][S] f16;  O: [B][S][D] f16.
// Double-buffered async staging of 64-key blocks; online softmax in f32;
// QK^T and P@V via WMMA.
// ---------------------------------------------------------------------------
__global__ __launch_bounds__(128) void attn_flash_kernel(
    const _Float16* __restrict__ Q, const _Float16* __restrict__ K,
    const _Float16* __restrict__ Vt, const int* __restrict__ valid_nums,
    _Float16* __restrict__ O)
{
    const int q0 = blockIdx.x * 64;
    const int bh = blockIdx.y;       // b*H + h
    const int b  = bh >> 4;
    const int h  = bh & 15;
    const int lane = threadIdx.x & 31;
    const int w    = threadIdx.x >> 5;
    const int l15  = lane & 15;
    const int hi   = (lane & 16) ? 1 : 0;
    const int koff  = hi ? 8 : 0;
    const int kbase = hi ? 16 : 0;
    const int rowb  = hi ? 8 : 0;
    const int valid = valid_nums[b];

    __shared__ _Float16 Ks[2][64][72];   // double-buffered 64 keys x 64 dh (+pad)
    __shared__ _Float16 Vs[2][64][72];   // double-buffered 64 dh x 64 keys (Vt tile)
    __shared__ _Float16 Ps[4][16][72];   // per-wave P tile 16 x 64

    const int ldr = threadIdx.x >> 1;        // 0..63
    const int ldc = (threadIdx.x & 1) * 32;  // 0 or 32 halfs

    // stage one 64-key block (8 async copies per thread: 64B K + 64B Vt)
    auto stage = [&](int j0, int bsel) {
        const _Float16* gk = K + (((size_t)bh * ATTN_S) + (j0 + ldr)) * ATTN_DH + ldc;
        _Float16* lk = &Ks[bsel][ldr][ldc];
        async_b128<0>(gk, lk);  async_b128<16>(gk, lk);
        async_b128<32>(gk, lk); async_b128<48>(gk, lk);
        const _Float16* gv = Vt + (((size_t)bh * ATTN_DH) + ldr) * ATTN_S + j0 + ldc;
        _Float16* lv = &Vs[bsel][ldr][ldc];
        async_b128<0>(gv, lv);  async_b128<16>(gv, lv);
        async_b128<32>(gv, lv); async_b128<48>(gv, lv);
    };

    // Q fragments for this wave's 16 rows (DH=64 -> two K=32 fragments)
    const _Float16* Qrow = Q + (((size_t)bh * ATTN_S) + (q0 + w * 16 + l15)) * ATTN_DH;
    v16h qf[2];
    for (int ks = 0; ks < 2; ++ks) {
        ((v8h*)&qf[ks])[0] = *(const v8h*)(Qrow + ks * 32 + koff);
        ((v8h*)&qf[ks])[1] = *(const v8h*)(Qrow + ks * 32 + koff + 16);
    }

    v8f oacc[4] = {};
    float mrun[8], lrun[8];
    for (int r = 0; r < 8; ++r) { mrun[r] = -__builtin_inff(); lrun[r] = 0.f; }

    stage(0, 0);   // prologue

    int buf = 0;
    for (int j0 = 0; j0 < valid; j0 += 64, buf ^= 1) {
        if (j0 + 64 < valid) {
            stage(j0 + 64, buf ^ 1);  // keep next block in flight
            wait_async_le<8>();       // retire current block (in-order)
        } else {
            wait_async_le<0>();
        }
        __syncthreads();

        // scores: 16 (q) x 64 (keys) per wave = 4 n-tiles, K-dim = DH = 64
        v8f sa[4] = {};
        for (int ks = 0; ks < 2; ++ks) {
            for (int t = 0; t < 4; ++t) {
                v16h bf;
                ((v8h*)&bf)[0] = *(const v8h*)&Ks[buf][t * 16 + l15][ks * 32 + koff];
                ((v8h*)&bf)[1] = *(const v8h*)&Ks[buf][t * 16 + l15][ks * 32 + koff + 16];
                sa[t] = __builtin_amdgcn_wmma_f32_16x16x32_f16(
                    false, qf[ks], false, bf, (short)0, sa[t], false, false);
            }
        }

        // scale + key-length mask
        float sc[4][8];
        for (int t = 0; t < 4; ++t) {
            const bool ok = (j0 + t * 16 + l15) < valid;
            for (int r = 0; r < 8; ++r)
                sc[t][r] = ok ? sa[t][r] * 0.125f : -__builtin_inff();
        }

        // online softmax per row (rows rowb+r live in this lane-half)
        for (int r = 0; r < 8; ++r) {
            float mb = fmaxf(fmaxf(sc[0][r], sc[1][r]), fmaxf(sc[2][r], sc[3][r]));
            for (int mm = 1; mm < 16; mm <<= 1) mb = fmaxf(mb, __shfl_xor(mb, mm, 32));
            const float mn   = fmaxf(mrun[r], mb);
            const float corr = __expf(mrun[r] - mn);
            float ps = 0.f;
            for (int t = 0; t < 4; ++t) {
                const float p = __expf(sc[t][r] - mn);
                ps += p;
                Ps[w][rowb + r][t * 16 + l15] = (_Float16)p;
            }
            for (int mm = 1; mm < 16; mm <<= 1) ps += __shfl_xor(ps, mm, 32);
            lrun[r] = lrun[r] * corr + ps;
            mrun[r] = mn;
            for (int t = 0; t < 4; ++t) oacc[t][r] *= corr;
        }
        __syncthreads();   // order Ps writes before fragment reads

        // O += P @ V  (K-dim = 64 keys, N = 64 dh)
        for (int ks = 0; ks < 2; ++ks) {
            v16h pf;
            ((v8h*)&pf)[0] = *(const v8h*)&Ps[w][l15][ks * 32 + koff];
            ((v8h*)&pf)[1] = *(const v8h*)&Ps[w][l15][ks * 32 + koff + 16];
            for (int t = 0; t < 4; ++t) {
                v16h bf;
                ((v8h*)&bf)[0] = *(const v8h*)&Vs[buf][t * 16 + l15][ks * 32 + koff];
                ((v8h*)&bf)[1] = *(const v8h*)&Vs[buf][t * 16 + l15][ks * 32 + koff + 16];
                oacc[t] = __builtin_amdgcn_wmma_f32_16x16x32_f16(
                    false, pf, false, bf, (short)0, oacc[t], false, false);
            }
        }
        __syncthreads();
    }

    // normalize and write O tile: f16 [B][S][D], d = h*64 + dh
    for (int t = 0; t < 4; ++t)
        for (int r = 0; r < 8; ++r) {
            const int sq = q0 + w * 16 + rowb + r;
            O[((size_t)(b * ATTN_S + sq)) * ATTN_D + h * 64 + t * 16 + l15] =
                (_Float16)(oacc[t][r] / lrun[r]);
        }
}

// ---------------------------------------------------------------------------
// Host-side launch
// ---------------------------------------------------------------------------
extern "C" void kernel_launch(void* const* d_in, const int* in_sizes, int n_in,
                              void* d_out, int out_size, void* d_ws, size_t ws_size,
                              hipStream_t stream) {
    (void)in_sizes; (void)n_in; (void)out_size; (void)ws_size;

    const float* x  = (const float*)d_in[0];
    const int*   vn = (const int*)  d_in[1];
    const float* Wq = (const float*)d_in[2];
    const float* bq = (const float*)d_in[3];
    const float* Wk = (const float*)d_in[4];
    const float* bk = (const float*)d_in[5];
    const float* Wv = (const float*)d_in[6];
    const float* bv = (const float*)d_in[7];
    const float* Wo = (const float*)d_in[8];
    const float* bo = (const float*)d_in[9];

    char* ws = (char*)d_ws;
    const size_t XH_BYTES = (size_t)ATTN_M * ATTN_D * sizeof(_Float16);   // 8 MB
    const size_t W_BYTES  = (size_t)ATTN_D * ATTN_D * sizeof(_Float16);   // 2 MB
    _Float16* xh  = (_Float16*)(ws);
    _Float16* Wqt = (_Float16*)(ws + XH_BYTES);
    _Float16* Wkt = (_Float16*)(ws + XH_BYTES + W_BYTES);
    _Float16* Wvt = (_Float16*)(ws + XH_BYTES + 2 * W_BYTES);
    _Float16* Wot = (_Float16*)(ws + XH_BYTES + 3 * W_BYTES);
    _Float16* Qb  = (_Float16*)(ws + XH_BYTES + 4 * W_BYTES);
    _Float16* Kb  = (_Float16*)(ws + 2 * XH_BYTES + 4 * W_BYTES);
    _Float16* Vtb = (_Float16*)(ws + 3 * XH_BYTES + 4 * W_BYTES);
    _Float16* Ob  = (_Float16*)(ws + 4 * XH_BYTES + 4 * W_BYTES);

    // 1) convert x to f16
    {
        const int n4 = ATTN_M * ATTN_D / 4;
        cvt_x_kernel<<<(n4 + 255) / 256, 256, 0, stream>>>(x, xh, n4);
    }
    // 2) convert + transpose weights
    {
        dim3 g(ATTN_D / 32, ATTN_D / 32), blk(32, 8);
        cvt_tr_kernel<<<g, blk, 0, stream>>>(Wq, Wqt);
        cvt_tr_kernel<<<g, blk, 0, stream>>>(Wk, Wkt);
        cvt_tr_kernel<<<g, blk, 0, stream>>>(Wv, Wvt);
        cvt_tr_kernel<<<g, blk, 0, stream>>>(Wo, Wot);
    }
    // 3) QKV projections
    {
        dim3 g(ATTN_M / 64, ATTN_D / 64);
        gemm_wmma_kernel<<<g, 128, 0, stream>>>(xh, Wqt, bq, (void*)Qb,  0);
        gemm_wmma_kernel<<<g, 128, 0, stream>>>(xh, Wkt, bk, (void*)Kb,  0);
        gemm_wmma_kernel<<<g, 128, 0, stream>>>(xh, Wvt, bv, (void*)Vtb, 1);
    }
    // 4) flash attention
    {
        dim3 g(ATTN_S / 64, ATTN_B * ATTN_H);
        attn_flash_kernel<<<g, 128, 0, stream>>>(Qb, Kb, Vtb, vn, Ob);
    }
    // 5) output projection -> fp32 d_out
    {
        dim3 g(ATTN_M / 64, ATTN_D / 64);
        gemm_wmma_kernel<<<g, 128, 0, stream>>>(Ob, Wot, bo, d_out, 3);
    }
}
